// GSCSNN_15831249453646
// MI455X (gfx1250) — matrix operand order repaired and checked by reading, not compiled, not run on hardware
//
#include <hip/hip_runtime.h>
#include <hip/hip_bf16.h>
#include <stdint.h>

// ---------------------------------------------------------------------------
// Adaptive-LIF recurrent SNN scan for MI455X (gfx1250).
//   B=256, T=200, N_IN=120, N_HID=1024, N_OUT=12, THRESHOLD=1.0
//
// Latency-bound recurrence (total math ~120 GFLOP, unique bytes ~30 MB, HBM
// floor ~1.3us -> the 200-step dependency chain dominates):
//  * persistent kernel, 64 WGs x 256 thr (8 wave32 waves), global sense barrier
//  * Wrec/W1/Wout pre-swizzled into WMMA B-fragments in LDS (HBM read once)
//  * spikes are exactly {0,1} -> exact in bf16; f32 weights split hi/lo bf16
//    so spk@W.T == 2x v_wmma_f32_16x16x32_bf16 with ~fp32 accuracy
//  * x pre-split/pre-swizzled into A-fragments by a parallel prep kernel
//  * readout recurrence is linear: out_sum = sum_s (1-beta^(T-s)) * p_s,
//    folded per-step with 4 peeled WMMAs; deterministic final reduction
//  * K-loop fully unrolled with hoisted base pointers -> immediate-offset
//    global_load_b128 / ds_load_b128 feeding v_wmma, no per-iter address ALU
// ---------------------------------------------------------------------------

typedef __attribute__((ext_vector_type(16))) __bf16 v16bf;
typedef __attribute__((ext_vector_type(8)))  float  v8f;
typedef unsigned short u16;

#define BB    256
#define TT    200
#define NIN   120
#define NHID  1024
#define NOUT  12

#define NWG   64
#define NTH   256

// workspace layout (bytes)
#define SPK_ELEMS ((size_t)32*16*512)                 // u16 per ping buffer
#define XF_ELEMS  ((size_t)TT*16*4*2*512)             // x fragments (hi+lo)
#define OFF_SPK   ((size_t)0)
#define OFF_XF    ((size_t)2*SPK_ELEMS*sizeof(u16))   // 1 MiB
#define OFF_PART  (OFF_XF + XF_ELEMS*sizeof(u16))
#define PART_ELEMS ((size_t)NWG*BB*16)
#define OFF_CNT   (OFF_PART + PART_ELEMS*sizeof(float))
// total ~28.3 MB of d_ws

static __device__ __forceinline__ void bf16_split(float v, u16& hi, u16& lo) {
  __bf16 h = (__bf16)v;
  float hf = (float)h;
  __bf16 l = (__bf16)(v - hf);
  hi = __builtin_bit_cast(u16, h);
  lo = __builtin_bit_cast(u16, l);
}
static __device__ __forceinline__ v8f vzero() {
  v8f z;
#pragma unroll
  for (int i = 0; i < 8; ++i) z[i] = 0.0f;
  return z;
}
static __device__ __forceinline__ v16bf ldfrag(const u16* p) {
  return *(const v16bf*)p;
}
static __device__ __forceinline__ v8f wmma_bf16(v16bf a, v16bf b, v8f c) {
  // D = A(16x32) * B(32x16) + C, f32 accumulate
  return __builtin_amdgcn_wmma_f32_16x16x32_bf16(false, a, false, b,
                                                 (short)0, c, false, false);
}

// ---------------------------------------------------------------------------
// Prep: split x into hi/lo bf16 and store in WMMA A-fragment layout:
//   A-layout: lane L (M = L%16, half = L/16), elem i: K = 16*(i/8)+8*half+(i%8)
//   fragment (t, mtile, kc) -> 512 u16 hi, then 512 u16 lo
// ---------------------------------------------------------------------------
__global__ void __launch_bounds__(NTH) snn_prep_xfrags(const float* __restrict__ x,
                                                       u16* __restrict__ xf) {
  int gid = blockIdx.x * NTH + threadIdx.x;
  if (gid >= TT * 16 * 4 * 32) return;
  int lane = gid & 31;
  int r = gid >> 5;
  int kc = r & 3;  r >>= 2;
  int mt = r & 15; r >>= 4;
  int t = r;
  int b = mt * 16 + (lane & 15);
  int half = lane >> 4;
  size_t base = ((size_t)((t * 16 + mt) * 4 + kc)) * 2 * 512 + (size_t)lane * 16;
#pragma unroll
  for (int i = 0; i < 16; ++i) {
    int kk = kc * 32 + 16 * (i >> 3) + 8 * half + (i & 7);
    float v = (kk < NIN) ? x[((size_t)b * TT + t) * NIN + kk] : 0.0f;
    u16 hi, lo; bf16_split(v, hi, lo);
    xf[base + i]       = hi;
    xf[base + 512 + i] = lo;
  }
}

// ---------------------------------------------------------------------------
// Persistent recurrent kernel. WG w owns hidden cols [16w,16w+16);
// wave v owns batch rows [32v,32v+32) = M-tiles 2v, 2v+1.
// B-fragment layout assumed: lane L (N = L%16), elem i: K = 16*(i/8) + 8*(L/16) + (i%8).
// ---------------------------------------------------------------------------
__global__ void __launch_bounds__(NTH) snn_recurrent(
    const float* __restrict__ W1, const float* __restrict__ Wrec,
    const float* __restrict__ Wout, const float* __restrict__ alpha,
    const float* __restrict__ rho, const float* __restrict__ beta_a,
    const float* __restrict__ beta_o, const u16* __restrict__ xf,
    u16* __restrict__ spk, float* __restrict__ part,
    unsigned* __restrict__ cnt, float* __restrict__ out) {

  __shared__ __attribute__((aligned(32))) u16 sWrec[32 * 2 * 512];  // 64 KB
  __shared__ __attribute__((aligned(32))) u16 sW1[4 * 2 * 512];     // 8 KB
  __shared__ __attribute__((aligned(32))) u16 sWout[2 * 512];       // 2 KB

  const int wg = blockIdx.x;
  const int tid = threadIdx.x;
  const int wave = tid >> 5;
  const int lane = tid & 31;
  const int n = lane & 15;          // output column within tile == hidden col
  const int halfL = lane >> 4;
  const int mt0 = 2 * wave, mt1 = 2 * wave + 1;
  const int hbase = wg * 16;

  // ---- build LDS B-fragments (hi/lo bf16 split of f32 weights) ----
  for (int e = tid; e < 32 * 512; e += NTH) {
    int kc = e >> 9, rem = e & 511;
    int l = rem >> 4, i = rem & 15;
    int col = l & 15, krow = 16 * (i >> 3) + 8 * (l >> 4) + (i & 7);
    float v = Wrec[(size_t)(hbase + col) * NHID + (kc * 32 + krow)];
    u16 hi, lo; bf16_split(v, hi, lo);
    sWrec[(kc * 2 + 0) * 512 + rem] = hi;
    sWrec[(kc * 2 + 1) * 512 + rem] = lo;
  }
  for (int e = tid; e < 4 * 512; e += NTH) {
    int kc = e >> 9, rem = e & 511;
    int l = rem >> 4, i = rem & 15;
    int col = l & 15, krow = 16 * (i >> 3) + 8 * (l >> 4) + (i & 7);
    int kk = kc * 32 + krow;
    float v = (kk < NIN) ? W1[(size_t)(hbase + col) * NIN + kk] : 0.0f;
    u16 hi, lo; bf16_split(v, hi, lo);
    sW1[(kc * 2 + 0) * 512 + rem] = hi;
    sW1[(kc * 2 + 1) * 512 + rem] = lo;
  }
  for (int e = tid; e < 512; e += NTH) {
    int l = e >> 4, i = e & 15;
    int col = l & 15, krow = 16 * (i >> 3) + 8 * (l >> 4) + (i & 7);
    bool own = ((krow >> 4) == (wg & 1));  // mask to this WG's 16 columns
    float v = (own && col < NOUT) ? Wout[(size_t)col * NHID + ((wg >> 1) * 32 + krow)]
                                  : 0.0f;
    u16 hi, lo; bf16_split(v, hi, lo);
    sWout[e] = hi;
    sWout[512 + e] = lo;
  }
  __syncthreads();

  // per-lane cell parameters: every element a lane owns shares hidden col n
  const int h = hbase + n;
  const float aL  = alpha[h];
  const float omA = 1.0f - aL;
  const float rL  = rho[h];
  const float baL = beta_a[h];
  const float bo  = beta_o[n < NOUT ? n : 0];
  const float invBo = 1.0f / bo;
  float wpow = powf(bo, (float)(TT + 1));  // beta^(T-s), s = t-1 at t = 0

  const v16bf Woh = ldfrag(&sWout[(size_t)lane * 16]);
  const v16bf Wol = ldfrag(&sWout[512 + (size_t)lane * 16]);

  // hoisted per-lane LDS bases (immediate offsets inside the unrolled loops)
  const u16* sWp  = &sWrec[(size_t)lane * 16];
  const u16* sW1p = &sW1[(size_t)lane * 16];

  v8f v1a = vzero(), v1b = vzero();   // membrane
  v8f a1a = vzero(), a1b = vzero();   // adaptation
  v8f spa = vzero(), spb = vzero();   // previous spikes
  v8f acc0 = vzero(), acc1 = vzero(); // readout accumulator

  const int kcR = wg >> 1;                       // K-chunk holding own cols
  const int iP = 8 * (wg & 1) + (n & 7);         // A-layout elem for writes
  const int lpBase = 16 * (n >> 3) + 8 * halfL;  // A-layout lane (+r)
  unsigned epoch = 0;

  auto gbar = [&]() {
    __threadfence();
    __syncthreads();
    if (tid == 0) {
      __hip_atomic_fetch_add(cnt, 1u, __ATOMIC_RELAXED, __HIP_MEMORY_SCOPE_AGENT);
      ++epoch;
      while (__hip_atomic_load(cnt, __ATOMIC_RELAXED, __HIP_MEMORY_SCOPE_AGENT) <
             (unsigned)NWG * epoch)
        __builtin_amdgcn_s_sleep(1);
    }
    __syncthreads();
    __threadfence();
  };

  for (int t = 0; t < TT; ++t) {
    const u16* rbuf = spk + (size_t)(t & 1) * SPK_ELEMS;        // spk_{t-1}
    u16* wbuf = spk + (size_t)((t + 1) & 1) * SPK_ELEMS;        // spk_t

    // per-lane global bases; K-loop offsets are compile-time immediates
    const u16* pA0 = rbuf + (size_t)mt0 * 512 + lane * 16;
    const u16* pA1 = rbuf + (size_t)mt1 * 512 + lane * 16;

    v8f c0 = vzero(), c1 = vzero();

    // I1 = spk_{t-1} @ Wrec.T  (hi + lo splits) -- branch-free hot loop
#pragma unroll
    for (int kc = 0; kc < 32; ++kc) {
      v16bf A0 = ldfrag(pA0 + kc * 8192);          // frag stride 16*512 u16
      v16bf A1 = ldfrag(pA1 + kc * 8192);
      v16bf Bh = ldfrag(sWp + kc * 1024);
      v16bf Bl = ldfrag(sWp + kc * 1024 + 512);
      c0 = wmma_bf16(A0, Bh, c0);
      c1 = wmma_bf16(A1, Bh, c1);
      c0 = wmma_bf16(A0, Bl, c0);
      c1 = wmma_bf16(A1, Bl, c1);
    }

    // peeled readout partial p_{t-1} over this WG's 16 cols (A frags L0-hot)
    v8f q0 = vzero(), q1 = vzero();
    {
      v16bf A0 = ldfrag(pA0 + kcR * 8192);
      v16bf A1 = ldfrag(pA1 + kcR * 8192);
      q0 = wmma_bf16(A0, Woh, q0); q0 = wmma_bf16(A0, Wol, q0);
      q1 = wmma_bf16(A1, Woh, q1); q1 = wmma_bf16(A1, Wol, q1);
    }

    // I1 += x_t @ W1.T  (xh*Wh + xh*Wl + xl*Wh)
    const u16* pX0 = xf + (size_t)(t * 16 + mt0) * 4096 + lane * 16;
    const u16* pX1 = xf + (size_t)(t * 16 + mt1) * 4096 + lane * 16;
#pragma unroll
    for (int kc = 0; kc < 4; ++kc) {
      v16bf Xh0 = ldfrag(pX0 + kc * 1024), Xl0 = ldfrag(pX0 + kc * 1024 + 512);
      v16bf Xh1 = ldfrag(pX1 + kc * 1024), Xl1 = ldfrag(pX1 + kc * 1024 + 512);
      v16bf Wh = ldfrag(sW1p + kc * 1024);
      v16bf Wl = ldfrag(sW1p + kc * 1024 + 512);
      c0 = wmma_bf16(Xh0, Wh, c0);
      c1 = wmma_bf16(Xh1, Wh, c1);
      c0 = wmma_bf16(Xh0, Wl, c0);
      c1 = wmma_bf16(Xh1, Wl, c1);
      c0 = wmma_bf16(Xl0, Wh, c0);
      c1 = wmma_bf16(Xl1, Wh, c1);
    }

    // fold readout partial for s = t-1 with weight (1 - beta^(T-s))
    float coef = 1.0f - wpow;
    wpow *= invBo;
#pragma unroll
    for (int r = 0; r < 8; ++r) { acc0[r] += coef * q0[r]; acc1[r] += coef * q1[r]; }

    // adaptive-LIF update + spike write (A-fragment layout for next step)
    size_t w0 = (size_t)(kcR * 16 + mt0) * 512;
    size_t w1 = (size_t)(kcR * 16 + mt1) * 512;
#pragma unroll
    for (int r = 0; r < 8; ++r) {
      float nv = aL * (v1a[r] - spa[r]) + omA * (c0[r] - a1a[r]);
      float s  = (nv > 1.0f) ? 1.0f : 0.0f;
      a1a[r] = rL * a1a[r] + baL * s;
      v1a[r] = nv; spa[r] = s;
      wbuf[w0 + (size_t)(lpBase + r) * 16 + iP] = (s != 0.0f) ? (u16)0x3F80 : (u16)0;

      float nv1 = aL * (v1b[r] - spb[r]) + omA * (c1[r] - a1b[r]);
      float s1  = (nv1 > 1.0f) ? 1.0f : 0.0f;
      a1b[r] = rL * a1b[r] + baL * s1;
      v1b[r] = nv1; spb[r] = s1;
      wbuf[w1 + (size_t)(lpBase + r) * 16 + iP] = (s1 != 0.0f) ? (u16)0x3F80 : (u16)0;
    }

    gbar();
  }

  // final readout term: s = T-1, spikes sit in buffer (TT & 1) == 0
  {
    const u16* rbuf = spk + (size_t)(TT & 1) * SPK_ELEMS;
    v16bf A0 = ldfrag(rbuf + ((size_t)(kcR * 16 + mt0) * 512 + lane * 16));
    v16bf A1 = ldfrag(rbuf + ((size_t)(kcR * 16 + mt1) * 512 + lane * 16));
    v8f q0 = vzero(), q1 = vzero();
    q0 = wmma_bf16(A0, Woh, q0); q0 = wmma_bf16(A0, Wol, q0);
    q1 = wmma_bf16(A1, Woh, q1); q1 = wmma_bf16(A1, Wol, q1);
    float coef = 1.0f - wpow;  // == 1 - beta
#pragma unroll
    for (int r = 0; r < 8; ++r) { acc0[r] += coef * q0[r]; acc1[r] += coef * q1[r]; }
  }

  // per-WG partials [wg][b][16]; deterministic fixed-order reduce by WG 0
#pragma unroll
  for (int r = 0; r < 8; ++r) {
    int b0 = mt0 * 16 + r + 8 * halfL;
    int b1 = mt1 * 16 + r + 8 * halfL;
    part[((size_t)wg * BB + b0) * 16 + n] = acc0[r];
    part[((size_t)wg * BB + b1) * 16 + n] = acc1[r];
  }
  gbar();

  if (wg == 0) {
    for (int e = tid; e < BB * NOUT; e += NTH) {
      int b = e / NOUT, o = e % NOUT;
      float s = 0.0f;
      for (int g = 0; g < NWG; ++g) s += part[((size_t)g * BB + b) * 16 + o];
      out[e] = s * (1.0f / TT);
    }
  }
}

// ---------------------------------------------------------------------------
extern "C" void kernel_launch(void* const* d_in, const int* in_sizes, int n_in,
                              void* d_out, int out_size, void* d_ws, size_t ws_size,
                              hipStream_t stream) {
  (void)in_sizes; (void)n_in; (void)out_size; (void)ws_size;
  const float* x      = (const float*)d_in[0];
  const float* W1     = (const float*)d_in[1];
  const float* Wrec   = (const float*)d_in[2];
  const float* Wout   = (const float*)d_in[3];
  const float* alpha  = (const float*)d_in[4];
  const float* rho    = (const float*)d_in[5];
  const float* beta_a = (const float*)d_in[6];
  const float* beta_o = (const float*)d_in[7];

  char* ws = (char*)d_ws;
  u16* spk      = (u16*)(ws + OFF_SPK);
  u16* xfrags   = (u16*)(ws + OFF_XF);
  float* part   = (float*)(ws + OFF_PART);
  unsigned* cnt = (unsigned*)(ws + OFF_CNT);

  // zero ping-buffer 0 (spk_{-1} == 0) and the barrier counter every call
  hipMemsetAsync(spk, 0, SPK_ELEMS * sizeof(u16), stream);
  hipMemsetAsync(cnt, 0, 64, stream);

  snn_prep_xfrags<<<(TT * 16 * 4 * 32 + NTH - 1) / NTH, NTH, 0, stream>>>(x, xfrags);
  snn_recurrent<<<NWG, NTH, 0, stream>>>(W1, Wrec, Wout, alpha, rho, beta_a, beta_o,
                                         xfrags, spk, part, cnt, (float*)d_out);
}
